// MambaBlock_8718783611649
// MI455X (gfx1250) — matrix operand order repaired and checked by reading, not compiled, not run on hardware
//
#include <hip/hip_runtime.h>
#include <hip/hip_bf16.h>
#include <math.h>

// Problem constants (match reference)
#define BSZ 8
#define LSEQ 4096
#define DM 256
#define SST 16
#define KC 4
#define BL (BSZ * LSEQ)          // 32768 rows
#define BLD ((size_t)BL * DM)    // 8,388,608
#define BLS ((size_t)BL * SST)   // 524,288
#define KBLKS (DM / 32)          // 8 K-blocks of 32

typedef __attribute__((ext_vector_type(16))) __bf16 v16bf;
typedef __attribute__((ext_vector_type(8)))  __bf16 v8bf;
typedef __attribute__((ext_vector_type(8)))  float  v8f;

// ---------------- WMMA fragment loaders (bf16 16x16x32, wave32 layouts) ------

// A matrix 16x32 (MxK) from a bf16 row-major source: two 16-byte loads, no cvt.
// lane l: row = row0 + (l&15); elements 0..7 -> K = k0 + 8*(l>>4) + i,
//         elements 8..15 -> K = k0 + 8*(l>>4) + 16 + (i-8)
__device__ __forceinline__ v16bf load_a_frag_bf(const __bf16* __restrict__ X,
                                                int ldx, int row0, int k0, int lane) {
    const int g = lane >> 4, lr = lane & 15;
    const __bf16* p = X + (size_t)(row0 + lr) * ldx + (k0 + 8 * g);
    const v8bf lo = *(const v8bf*)(p);
    const v8bf hi = *(const v8bf*)(p + 16);
    v16bf a;
#pragma unroll
    for (int i = 0; i < 8; ++i) { a[i] = lo[i]; a[8 + i] = hi[i]; }
    return a;
}

// B fragment from prepacked bf16 buffer: fragment f = ntile*kblocks + kb,
// each lane owns 16 contiguous bf16 (32 B) -> straight vector loads, no cvt.
__device__ __forceinline__ v16bf load_b_packed(const __bf16* __restrict__ P,
                                               int ntile, int kb, int kblocks, int lane) {
    const __bf16* p = P + (((size_t)(ntile * kblocks + kb) * 32 + lane) << 4);
    return *(const v16bf*)p;
}

__device__ __forceinline__ float silu_f(float v) {
    // v * sigmoid(v) with hardware rcp (no div_scale/Newton chain)
    return v * __builtin_amdgcn_rcpf(1.0f + __expf(-v));
}

// ---------------- Prologue: f32 -> bf16 row-major copy (4 elems/thread) ------
__global__ void k_cvt_bf16(const float* __restrict__ src, __bf16* __restrict__ dst) {
    const size_t i = ((size_t)blockIdx.x * blockDim.x + threadIdx.x) * 4;
    if (i >= BLD) return;
    const float4 v = *(const float4*)(src + i);
    dst[i + 0] = (__bf16)v.x; dst[i + 1] = (__bf16)v.y;
    dst[i + 2] = (__bf16)v.z; dst[i + 3] = (__bf16)v.w;
}

// ---------------- Weight prepack into per-lane fragment layout ---------------
// B[k][n] = W[n*strideN + k*strideK]; lane l covers col n = ntile*16 + (l&15),
// K = kb*32 + 8*(l>>4) + (i&7) + 16*(i>>3), i = 0..15 (matches A fragment order).
__global__ void k_pack_b(const float* __restrict__ W, int strideN, int strideK,
                         int ntiles, int kblocks, __bf16* __restrict__ out) {
    const int idx = blockIdx.x * blockDim.x + threadIdx.x;
    const int total = ntiles * kblocks * 32 * 16;
    if (idx >= total) return;
    const int i = idx & 15;
    const int lane = (idx >> 4) & 31;
    const int f = idx >> 9;
    const int nt = f / kblocks, kb = f - nt * kblocks;
    const int n = nt * 16 + (lane & 15);
    const int k = kb * 32 + 8 * (lane >> 4) + (i & 7) + 16 * (i >> 3);
    out[idx] = (__bf16)W[(size_t)n * strideN + (size_t)k * strideK];
}

// ---------------- Kernel 1: in_proj GEMM [BL,256]x[256,512], 32x64/wave ------
// ng<4 -> xp columns (0..255); ng>=4 -> silu(res) columns (256..511) [uniform]
__global__ void k_inproj(const __bf16* __restrict__ xbf, const __bf16* __restrict__ wpk,
                         const float* __restrict__ in_b,
                         float* __restrict__ xp, float* __restrict__ sres) {
    const int lane = threadIdx.x & 31;
    const int tile = blockIdx.x * 8 + (threadIdx.x >> 5); // 8192 macro tiles
    const int mg = tile >> 3;   // 1024 row groups (32 rows)
    const int ng = tile & 7;    // 8 col groups (64 cols, N = 512)
    const int row0 = mg * 32, nt0 = ng * 4;

    v8f c[2][4] = {};
    for (int kb = 0; kb < KBLKS; ++kb) {
        const int k0 = kb * 32;
        const v16bf a0 = load_a_frag_bf(xbf, DM, row0, k0, lane);
        const v16bf a1 = load_a_frag_bf(xbf, DM, row0 + 16, k0, lane);
#pragma unroll
        for (int j = 0; j < 4; ++j) {
            const v16bf b = load_b_packed(wpk, nt0 + j, kb, KBLKS, lane);
            c[0][j] = __builtin_amdgcn_wmma_f32_16x16x32_bf16(false, a0, false, b,
                                                              (short)0, c[0][j], false, false);
            c[1][j] = __builtin_amdgcn_wmma_f32_16x16x32_bf16(false, a1, false, b,
                                                              (short)0, c[1][j], false, false);
        }
    }
    const int g = lane >> 4, lr = lane & 15;
    if (ng < 4) {  // wave-uniform: all columns land in xp
#pragma unroll
        for (int j = 0; j < 4; ++j) {
            const int n = (nt0 + j) * 16 + lr;
            const float bias = in_b[n];
#pragma unroll
            for (int mi = 0; mi < 2; ++mi)
#pragma unroll
                for (int r = 0; r < 8; ++r) {
                    const size_t m = (size_t)row0 + mi * 16 + 8 * g + r;
                    xp[m * DM + n] = c[mi][j][r] + bias;
                }
        }
    } else {       // wave-uniform: all columns land in silu(res)
#pragma unroll
        for (int j = 0; j < 4; ++j) {
            const int n = (nt0 + j) * 16 + lr;
            const float bias = in_b[n];
#pragma unroll
            for (int mi = 0; mi < 2; ++mi)
#pragma unroll
                for (int r = 0; r < 8; ++r) {
                    const size_t m = (size_t)row0 + mi * 16 + 8 * g + r;
                    sres[m * DM + (n - DM)] = silu_f(c[mi][j][r] + bias);
                }
        }
    }
}

// ---------------- Kernel 2: depthwise causal conv1d (K=4), dual f32+bf16 out -
__global__ void k_conv(const float* __restrict__ xp, const float* __restrict__ conv_w,
                       const float* __restrict__ conv_b,
                       float* __restrict__ xconv, __bf16* __restrict__ xconv_bf) {
    const size_t idx = (size_t)blockIdx.x * blockDim.x + threadIdx.x;
    if (idx >= BLD) return;
    const int d = (int)(idx % DM);
    const int l = (int)((idx / DM) % LSEQ);
    const size_t row_base = idx - (size_t)d;
    float acc = conv_b[d];
#pragma unroll
    for (int j = 0; j < KC; ++j) {
        const int ls = l - (KC - 1) + j;
        if (ls >= 0) acc += xp[row_base + (size_t)(j - (KC - 1)) * DM] * conv_w[d * KC + j];
    }
    xconv[idx] = acc;
    xconv_bf[idx] = (__bf16)acc;
}

// ---------------- Kernel 3: dA = exp(xconv@dt), dB = xconv@B_w^T, 32x32/wave -
// packed B buffer: frags [0..7] = dt columns, frags [8..15] = B_w columns
__global__ void k_dadb(const __bf16* __restrict__ xconv_bf, const __bf16* __restrict__ wpk,
                       float* __restrict__ dA, float* __restrict__ dB) {
    const int lane = threadIdx.x & 31;
    const int mg = blockIdx.x * 8 + (threadIdx.x >> 5); // 1024 row groups
    const int row0 = mg * 32;

    v8f c[2][2] = {};
    for (int kb = 0; kb < KBLKS; ++kb) {
        const int k0 = kb * 32;
        const v16bf a0 = load_a_frag_bf(xconv_bf, DM, row0, k0, lane);
        const v16bf a1 = load_a_frag_bf(xconv_bf, DM, row0 + 16, k0, lane);
#pragma unroll
        for (int j = 0; j < 2; ++j) {  // j=0 -> dt (dA), j=1 -> B_w (dB)
            const v16bf b = load_b_packed(wpk, j, kb, KBLKS, lane);
            c[0][j] = __builtin_amdgcn_wmma_f32_16x16x32_bf16(false, a0, false, b,
                                                              (short)0, c[0][j], false, false);
            c[1][j] = __builtin_amdgcn_wmma_f32_16x16x32_bf16(false, a1, false, b,
                                                              (short)0, c[1][j], false, false);
        }
    }
    const int g = lane >> 4, lr = lane & 15; // lr = s index
#pragma unroll
    for (int mi = 0; mi < 2; ++mi)
#pragma unroll
        for (int r = 0; r < 8; ++r) {
            const size_t m = (size_t)row0 + mi * 16 + 8 * g + r;
            dA[m * SST + lr] = __expf(c[mi][0][r]);
            dB[m * SST + lr] = c[mi][1][r];
        }
}

// ---------------- Kernel 4: affine wave-scan over L per (b,s) chain ----------
// state (cp,yc): cp_t = cp*dA_t ; yc_t = yc + cp_t*dB_t ; y_t = yc_t - yc_{t-1}*dA_t
__global__ void k_scan(const float* __restrict__ dA, const float* __restrict__ dB,
                       float* __restrict__ y) {
    const int chain = blockIdx.x;       // 0..127 = b*16 + s
    const int b = chain >> 4, s = chain & 15;
    const int lane = threadIdx.x;       // wave32
    const size_t base = (size_t)b * LSEQ * SST + s;
    const int t0 = lane * (LSEQ / 32);  // 128 steps per lane

    float P = 1.0f, Q = 0.0f;
    for (int j = 0; j < LSEQ / 32; ++j) {
        const size_t idx = base + (size_t)(t0 + j) * SST;
        const float a = dA[idx], bb = dB[idx];
        P *= a; Q += P * bb;
    }
#pragma unroll
    for (int off = 1; off < 32; off <<= 1) {
        const float Pp = __shfl_up(P, off, 32);
        const float Qp = __shfl_up(Q, off, 32);
        if (lane >= off) { Q = Qp + Pp * Q; P = Pp * P; }
    }
    float cp = __shfl_up(P, 1, 32);
    float yc = __shfl_up(Q, 1, 32);
    if (lane == 0) { cp = 1.0f; yc = 0.0f; }
    for (int j = 0; j < LSEQ / 32; ++j) {
        const size_t idx = base + (size_t)(t0 + j) * SST;
        const float a = dA[idx], bb = dB[idx];
        cp *= a;
        const float ycp = yc;
        yc += cp * bb;
        y[idx] = ((t0 + j) == 0) ? bb : (yc - ycp * a);
    }
}

// ---------------- Kernel 5a: t1 = (y @ C_w^T) * silu(res), bf16 output -------
__global__ void k_yc(const float* __restrict__ y, const float* __restrict__ C_w,
                     const float* __restrict__ sres, __bf16* __restrict__ t1bf) {
    const size_t idx = (size_t)blockIdx.x * blockDim.x + threadIdx.x;
    if (idx >= BLD) return;
    const int d = (int)(idx % DM);
    const size_t m = idx / DM;
    const float* yr = y + m * SST;
    const float* cw = C_w + (size_t)d * SST;
    float acc = 0.0f;
#pragma unroll
    for (int s = 0; s < SST; ++s) acc += yr[s] * cw[s];
    t1bf[idx] = (__bf16)(acc * sres[idx]);
}

// ---------------- Kernel 6: out_proj GEMM + bias + residual, 32x64/wave ------
__global__ void k_outproj(const __bf16* __restrict__ t1bf, const __bf16* __restrict__ wpk,
                          const float* __restrict__ out_b, const float* __restrict__ xconv,
                          float* __restrict__ out) {
    const int lane = threadIdx.x & 31;
    const int tile = blockIdx.x * 8 + (threadIdx.x >> 5); // 4096 macro tiles
    const int mg = tile >> 2;  // 1024 row groups
    const int ng = tile & 3;   // 4 col groups (N = 256)
    const int row0 = mg * 32, nt0 = ng * 4;

    v8f c[2][4] = {};
    for (int kb = 0; kb < KBLKS; ++kb) {
        const int k0 = kb * 32;
        const v16bf a0 = load_a_frag_bf(t1bf, DM, row0, k0, lane);
        const v16bf a1 = load_a_frag_bf(t1bf, DM, row0 + 16, k0, lane);
#pragma unroll
        for (int j = 0; j < 4; ++j) {
            const v16bf b = load_b_packed(wpk, nt0 + j, kb, KBLKS, lane);
            c[0][j] = __builtin_amdgcn_wmma_f32_16x16x32_bf16(false, a0, false, b,
                                                              (short)0, c[0][j], false, false);
            c[1][j] = __builtin_amdgcn_wmma_f32_16x16x32_bf16(false, a1, false, b,
                                                              (short)0, c[1][j], false, false);
        }
    }
    const int g = lane >> 4, lr = lane & 15;
#pragma unroll
    for (int j = 0; j < 4; ++j) {
        const int n = (nt0 + j) * 16 + lr;
        const float bias = out_b[n];
#pragma unroll
        for (int mi = 0; mi < 2; ++mi)
#pragma unroll
            for (int r = 0; r < 8; ++r) {
                const size_t m = (size_t)row0 + mi * 16 + 8 * g + r;
                out[m * DM + n] = c[mi][j][r] + bias + xconv[m * DM + n];
            }
    }
}

// ---------------- BatchNorm ---------------------------------------------------
__global__ void k_zero(float* __restrict__ p, int n) {
    const int i = blockIdx.x * blockDim.x + threadIdx.x;
    if (i < n) p[i] = 0.0f;
}

__global__ void k_bnsum(const float* __restrict__ out, float* __restrict__ sums,
                        float* __restrict__ sumsq) {
    const int d = threadIdx.x;                 // channel
    const int slab = blockIdx.x;               // 256 slabs of 128 rows
    float s = 0.0f, s2 = 0.0f;
    for (int r = 0; r < BL / 256; ++r) {
        const float v = out[((size_t)slab * (BL / 256) + r) * DM + d];
        s += v; s2 += v * v;
    }
    atomicAdd(&sums[d], s);
    atomicAdd(&sumsq[d], s2);
}

__global__ void k_bnapply(float* __restrict__ out, const float* __restrict__ sums,
                          const float* __restrict__ sumsq, const float* __restrict__ gamma,
                          const float* __restrict__ beta) {
    const size_t idx = (size_t)blockIdx.x * blockDim.x + threadIdx.x;
    if (idx >= BLD) return;
    const int d = (int)(idx % DM);
    const float inv_n = 1.0f / (float)BL;
    const float mean = sums[d] * inv_n;
    const float var = sumsq[d] * inv_n - mean * mean;
    out[idx] = (out[idx] - mean) * rsqrtf(var + 1e-5f) * gamma[d] + beta[d];
}

// ---------------- Host launcher ----------------------------------------------
extern "C" void kernel_launch(void* const* d_in, const int* in_sizes, int n_in,
                              void* d_out, int out_size, void* d_ws, size_t ws_size,
                              hipStream_t stream) {
    (void)in_sizes; (void)n_in; (void)out_size; (void)ws_size;
    const float* x      = (const float*)d_in[0];
    const float* dt     = (const float*)d_in[1];
    const float* B_w    = (const float*)d_in[2];
    const float* C_w    = (const float*)d_in[3];
    const float* conv_w = (const float*)d_in[4];
    const float* conv_b = (const float*)d_in[5];
    const float* in_w   = (const float*)d_in[6];
    const float* in_b   = (const float*)d_in[7];
    const float* out_w  = (const float*)d_in[8];
    const float* out_b  = (const float*)d_in[9];
    const float* gamma  = (const float*)d_in[10];
    const float* beta   = (const float*)d_in[11];
    float* out = (float*)d_out;

    float* ws = (float*)d_ws;
    float* xp    = ws;                 // BLD f32
    float* sres  = xp + BLD;           // BLD f32
    float* xconv = sres + BLD;         // BLD f32
    float* dA    = xconv + BLD;        // BLS f32
    float* dB    = dA + BLS;           // BLS f32
    float* y     = dB + BLS;           // BLS f32
    float* sums  = y + BLS;            // 256
    float* sumsq = sums + DM;          // 256

    // prepacked bf16 weight fragments (offsets stay 32B-aligned)
    __bf16* wpk_in   = (__bf16*)(sumsq + DM);           // 32*8*512 = 131072 bf16
    __bf16* wpk_out  = wpk_in + 32 * KBLKS * 512;       // 16*8*512 =  65536 bf16
    __bf16* wpk_dadb = wpk_out + 16 * KBLKS * 512;      //  2*8*512 =   8192 bf16
    // bf16 activation copies for GEMM A operands
    __bf16* xbf      = wpk_dadb + 2 * KBLKS * 512;      // BLD bf16
    __bf16* xconv_bf = xbf + BLD;                       // BLD bf16
    __bf16* t1bf     = xbf;                             // reuse xbf after in_proj

    const int elemBlocks = (int)((BLD + 255) / 256);

    // 0) prologue: pack weights; convert x to bf16
    k_pack_b<<<(32 * KBLKS * 512 + 255) / 256, 256, 0, stream>>>(in_w, DM, 1, 32, KBLKS, wpk_in);
    k_pack_b<<<(16 * KBLKS * 512 + 255) / 256, 256, 0, stream>>>(out_w, DM, 1, 16, KBLKS, wpk_out);
    k_pack_b<<<(1 * KBLKS * 512 + 255) / 256, 256, 0, stream>>>(dt, 1, SST, 1, KBLKS, wpk_dadb);
    k_pack_b<<<(1 * KBLKS * 512 + 255) / 256, 256, 0, stream>>>(B_w, DM, 1, 1, KBLKS,
                                                                wpk_dadb + KBLKS * 512);
    k_cvt_bf16<<<(int)((BLD / 4 + 255) / 256), 256, 0, stream>>>(x, xbf);
    // 1) in_proj: 1024x8 macro tiles (32x64), 8 waves/block
    k_inproj<<<(1024 * 8) / 8, 256, 0, stream>>>(xbf, wpk_in, in_b, xp, sres);
    // 2) depthwise causal conv (writes f32 + bf16)
    k_conv<<<elemBlocks, 256, 0, stream>>>(xp, conv_w, conv_b, xconv, xconv_bf);
    // 3) dA/dB: 1024 macro tiles (32 rows x both outputs)
    k_dadb<<<1024 / 8, 256, 0, stream>>>(xconv_bf, wpk_dadb, dA, dB);
    // 4) scan: one wave32 per (b,s) chain
    k_scan<<<BSZ * SST, 32, 0, stream>>>(dA, dB, y);
    // 5) y @ C_w^T * silu(res) -> bf16 (feeds out_proj A; overwrites xbf region)
    k_yc<<<elemBlocks, 256, 0, stream>>>(y, C_w, sres, t1bf);
    // 6) out_proj + bias + residual: 1024x4 macro tiles
    k_outproj<<<(1024 * 4) / 8, 256, 0, stream>>>(t1bf, wpk_out, out_b, xconv, out);
    // 7) BatchNorm
    k_zero<<<2, 256, 0, stream>>>(sums, 2 * DM);
    k_bnsum<<<256, 256, 0, stream>>>(out, sums, sumsq);
    k_bnapply<<<elemBlocks, 256, 0, stream>>>(out, sums, sumsq, gamma, beta);
}
